// AddAttention_84799834292629
// MI455X (gfx1250) — matrix operand order repaired and checked by reading, compiled, not run on hardware
//
#include <hip/hip_runtime.h>
#include <hip/hip_bf16.h>
#include <math.h>

#define BATCH 32
#define LCTX  4096
#define DIMD  256
#define MASK_FILL -1e25f

typedef __attribute__((ext_vector_type(16))) __bf16 v16bf;
typedef __attribute__((ext_vector_type(8)))  float  v8f;

// LDS row stride (in bf16 elems) for transposed Wc: 264*2 = 528 bytes,
// multiple of 16 bytes (b128-aligned rows), 528/4 = 132 banks -> 132%64 = 4
// bank offset per successive column: conflict-light.
#define WC_STRIDE 264

// fast tanh: 1 - 2*rcp(exp(2x)+1); v_exp_f32 + v_rcp_f32 (TRANS) + fma.
// Saturates correctly: exp->inf => rcp->0 => +1; exp->0 => -1.
__device__ __forceinline__ float fast_tanh(float x) {
    return fmaf(-2.0f, __builtin_amdgcn_rcpf(__expf(2.0f * x) + 1.0f), 1.0f);
}

// -----------------------------------------------------------------------------
// Kernel 1: scores[b,l] = sum_n tanh(qW[b,n] + (context @ Wc)[b,l,n]) * Ws[n]
// One block = 1 batch slice of 128 rows (8 waves x 16 rows/wave).
// GEMM via v_wmma_f32_16x16x32_bf16 with TWO independent accumulator chains
// (even/odd K-chunks) for intra-wave WMMA ILP. A (context rows) resident in
// VGPRs, B (Wc^T, bf16) staged in LDS.
// -----------------------------------------------------------------------------
__global__ __launch_bounds__(256)
void score_kernel(const float* __restrict__ query,
                  const float* __restrict__ context,
                  const float* __restrict__ Wq,
                  const float* __restrict__ Wc,
                  const float* __restrict__ Ws,
                  float* __restrict__ score)   // [B, L] region of d_out
{
    __shared__ __bf16 sWc[DIMD * WC_STRIDE];   // Wc transposed: sWc[n][k], 132 KB
    __shared__ float  sQW[DIMD];
    __shared__ float  sWs[DIMD];

    const int b  = blockIdx.y;
    const int m0 = blockIdx.x * 128;
    const int t  = threadIdx.x;

    // ---- stage Wc (f32, k-major) -> LDS as bf16, n-major (transposed) ----
    for (int idx = t; idx < DIMD * DIMD; idx += 256) {
        const int k = idx >> 8;        // row of Wc
        const int n = idx & 255;       // col of Wc (consecutive threads -> coalesced)
        sWc[n * WC_STRIDE + k] = (__bf16)Wc[idx];
    }

    // ---- qW[n] = sum_k q[b,k] * Wq[k,n]  (thread t owns column t) ----
    {
        float accq = 0.0f;
        const float* qb = query + (size_t)b * DIMD;
        #pragma unroll 4
        for (int k = 0; k < DIMD; ++k)
            accq = fmaf(qb[k], Wq[k * DIMD + t], accq);
        sQW[t] = accq;
        sWs[t] = Ws[t];
    }
    __syncthreads();

    const int wave  = t >> 5;
    const int lane  = t & 31;
    const int lrow  = lane & 15;           // M row within tile (both halves)
    const int koff  = (lane >> 4) * 16;    // A-fragment K half per lane half
    const int khalf = (lane >> 4) * 16;    // B-fragment K half per lane half
    const int col   = lane & 15;           // N column within tile

    const int mrow = m0 + wave * 16 + lrow;
    const float* arow = context + ((size_t)b * LCTX + mrow) * DIMD;

    // ---- load all A fragments (16 rows x K=256) as bf16, contiguous K order (k = 16h + s) ----
    v16bf afrag[8];
    #pragma unroll
    for (int kc = 0; kc < 8; ++kc) {
        const float* p = arow + kc * 32 + koff;
        #pragma unroll
        for (int e = 0; e < 8; ++e) {
            afrag[kc][e]     = (__bf16)p[e];        // K = 32*kc + koff + e
            afrag[kc][e + 8] = (__bf16)p[8 + e];    // K = 32*kc + koff + 8 + e
        }
    }

    float sacc[8];
    #pragma unroll
    for (int r = 0; r < 8; ++r) sacc[r] = 0.0f;

    // ---- N in chunks of 16, K in chunks of 32 (two independent WMMA chains) --
    for (int nc = 0; nc < 16; ++nc) {
        v8f acc0 = {};
        v8f acc1 = {};
        const int n = nc * 16 + col;
        const __bf16* bp = &sWc[n * WC_STRIDE + khalf];
        #pragma unroll
        for (int kc = 0; kc < 8; kc += 2) {
            v16bf bf0, bf1;
            #pragma unroll
            for (int e = 0; e < 16; ++e) {
                bf0[e] = bp[kc * 32 + e];            // K = 32*kc + khalf + e
                bf1[e] = bp[(kc + 1) * 32 + e];
            }
            acc0 = __builtin_amdgcn_wmma_f32_16x16x32_bf16(
                false, afrag[kc],     false, bf0, (short)0, acc0, false, false);
            acc1 = __builtin_amdgcn_wmma_f32_16x16x32_bf16(
                false, afrag[kc + 1], false, bf1, (short)0, acc1, false, false);
        }
        // acc[r] = h[m0 + wave*16 + r + (lane>=16 ? 8:0), n]
        const float qw = sQW[n];
        const float ws = sWs[n];
        #pragma unroll
        for (int r = 0; r < 8; ++r) {
            const float x = (acc0[r] + acc1[r]) + qw;
            sacc[r] = fmaf(fast_tanh(x), ws, sacc[r]);
        }
    }

    // ---- reduce over the 16 columns held by each lane-half ----
    #pragma unroll
    for (int r = 0; r < 8; ++r) {
        float v = sacc[r];
        v += __shfl_xor(v, 1, 32);
        v += __shfl_xor(v, 2, 32);
        v += __shfl_xor(v, 4, 32);
        v += __shfl_xor(v, 8, 32);
        sacc[r] = v;
    }
    if ((lane & 15) == 0) {
        const int rbase = m0 + wave * 16 + (lane >> 4) * 8;
        #pragma unroll
        for (int r = 0; r < 8; ++r)
            score[(size_t)b * LCTX + rbase + r] = sacc[r];
    }
}

// -----------------------------------------------------------------------------
// Kernel 2: masked softmax over L, in place (score -> weight). Zeroes z region.
// One block per batch, 256 threads x 16 strided elements.
// -----------------------------------------------------------------------------
__global__ __launch_bounds__(256)
void softmax_kernel(const int* __restrict__ mask,
                    float* __restrict__ w,     // [B, L] in/out
                    float* __restrict__ z)     // [B, D] zeroed here
{
    __shared__ float red[8];
    const int b = blockIdx.x;
    const int t = threadIdx.x;
    const int wave = t >> 5, lane = t & 31;

    float vals[16];
    float m = -3.0e38f;
    #pragma unroll
    for (int i = 0; i < 16; ++i) {
        const int l = i * 256 + t;
        float s = w[(size_t)b * LCTX + l];
        s = (mask[(size_t)b * LCTX + l] == 1) ? s : MASK_FILL;
        vals[i] = s;
        m = fmaxf(m, s);
    }
    #pragma unroll
    for (int d = 1; d < 32; d <<= 1) m = fmaxf(m, __shfl_xor(m, d, 32));
    if (lane == 0) red[wave] = m;
    __syncthreads();
    float mm = red[0];
    #pragma unroll
    for (int i = 1; i < 8; ++i) mm = fmaxf(mm, red[i]);
    __syncthreads();

    float sum = 0.0f;
    #pragma unroll
    for (int i = 0; i < 16; ++i) {
        const float e = __expf(vals[i] - mm);
        vals[i] = e;
        sum += e;
    }
    #pragma unroll
    for (int d = 1; d < 32; d <<= 1) sum += __shfl_xor(sum, d, 32);
    if (lane == 0) red[wave] = sum;
    __syncthreads();
    float tot = 0.0f;
    #pragma unroll
    for (int i = 0; i < 8; ++i) tot += red[i];

    const float inv = 1.0f / tot;
    #pragma unroll
    for (int i = 0; i < 16; ++i)
        w[(size_t)b * LCTX + i * 256 + t] = vals[i] * inv;

    z[b * DIMD + t] = 0.0f;   // pre-zero for kernel 3's atomics
}

// -----------------------------------------------------------------------------
// Kernel 3: z[b,d] = sum_l weight[b,l] * context[b,l,d]
// grid (B, 8): each block handles 512 rows; coalesced over d; atomic accumulate.
// -----------------------------------------------------------------------------
__global__ __launch_bounds__(256)
void wsum_kernel(const float* __restrict__ context,
                 const float* __restrict__ w,
                 float* __restrict__ z)
{
    __shared__ float sw[512];
    const int b = blockIdx.x;
    const int l0 = blockIdx.y * 512;
    const int t = threadIdx.x;

    for (int i = t; i < 512; i += 256)
        sw[i] = w[(size_t)b * LCTX + l0 + i];
    __syncthreads();

    const float* cp = context + ((size_t)b * LCTX + l0) * DIMD + t;
    float acc = 0.0f;
    #pragma unroll 4
    for (int i = 0; i < 512; ++i)
        acc = fmaf(sw[i], cp[(size_t)i * DIMD], acc);

    atomicAdd(&z[b * DIMD + t], acc);
}

// -----------------------------------------------------------------------------
extern "C" void kernel_launch(void* const* d_in, const int* in_sizes, int n_in,
                              void* d_out, int out_size, void* d_ws, size_t ws_size,
                              hipStream_t stream)
{
    (void)in_sizes; (void)n_in; (void)out_size; (void)d_ws; (void)ws_size;
    const float* query   = (const float*)d_in[0];
    const float* context = (const float*)d_in[1];
    const int*   cmask   = (const int*)  d_in[2];
    const float* Wq      = (const float*)d_in[3];
    const float* Wc      = (const float*)d_in[4];
    const float* Ws      = (const float*)d_in[5];

    float* weight = (float*)d_out;                       // [B, L]
    float* zvec   = weight + (size_t)BATCH * LCTX;       // [B, D]

    // 1) scores (heavy bf16-WMMA GEMM + tanh + Ws reduce), written into weight region
    score_kernel<<<dim3(LCTX / 128, BATCH), 256, 0, stream>>>(
        query, context, Wq, Wc, Ws, weight);
    // 2) masked softmax in place, zero z
    softmax_kernel<<<BATCH, 256, 0, stream>>>(cmask, weight, zvec);
    // 3) z = weight^T context
    wsum_kernel<<<dim3(BATCH, 8), 256, 0, stream>>>(context, weight, zvec);
}
